// NodeEarlyInteractionAdding_80539226734942
// MI455X (gfx1250) — matrix operand order repaired and checked by reading, not compiled, not run on hardware
//
#include <hip/hip_runtime.h>
#include <hip/hip_bf16.h>

typedef __bf16 bf16;
typedef bf16  v16bf __attribute__((ext_vector_type(16)));
typedef float v8f   __attribute__((ext_vector_type(8)));

static constexpr int kB = 1024;
static constexpr int kN = 36864;    // B * 36 nodes
static constexpr int kE = 138240;   // B * 135 edges

// ---------------------------------------------------------------------------
// Fused GEMM:  out = act( concat_segments(A) @ Bt^T + bias )
//   - A built from NSEG segments, each 128 cols wide (K = 128*NSEG); each
//     segment: base ptr (bf16 or f32), optional row-gather index, row stride.
//   - Bt = weight pre-transposed to (Nout x K) bf16.
//   - B tile (128 cols x 128 K-slab) staged into LDS per segment with
//     GLOBAL_LOAD_ASYNC_TO_LDS_B128 (ASYNCcnt-tracked DMA, no VGPR roundtrip),
//     shared by all 8 waves; rows padded +8 elems -> conflict-free ds_loads.
//   - Wave tile 32x64 (8 accumulators, 8 WMMAs/k-step); block tile 128x128.
//   - Epilogue: +bias, optional relu, then bf16 store OR f32 atomic scatter
//     (segment-sum) scaled by mult[row].
// ---------------------------------------------------------------------------

static constexpr int kLdsPad = 8;
static constexpr int kLdsLd  = 128 + kLdsPad;   // LDS row stride (elements)

__device__ __forceinline__ v16bf load_a_direct(const void* ptr, int ld, int isF32,
                                               int sr, int kc, int hf) {
  union { v16bf v; float4 f4[2]; bf16 e[16]; } a;
  if (!isF32) {
    const bf16* p = (const bf16*)ptr + (size_t)sr * ld + kc;
    a.f4[0] = *(const float4*)(p + hf * 8);
    a.f4[1] = *(const float4*)(p + 16 + hf * 8);
  } else {
    const float* p = (const float*)ptr + (size_t)sr * ld + kc;
#pragma unroll
    for (int j = 0; j < 8; ++j) {
      a.e[j]     = (bf16)p[hf * 8 + j];
      a.e[8 + j] = (bf16)p[16 + hf * 8 + j];
    }
  }
  return a.v;
}

__device__ __forceinline__ v16bf load_b_lds(const bf16* ldsb, int nl, int kc, int hf) {
  union { v16bf v; float4 f4[2]; } b;
  const bf16* p = ldsb + nl * kLdsLd + kc + hf * 8;
  b.f4[0] = *(const float4*)p;
  b.f4[1] = *(const float4*)(p + 16);
  return b.v;
}

__device__ __forceinline__ void epilogue_tile(v8f acc, int row0, int hf, int col,
                                              int M, const float* bias, int do_relu,
                                              bf16* outb, int out_ld, int out_off,
                                              float* aout, const int* sidx,
                                              const float* mult) {
  float bc = bias ? bias[col] : 0.f;
#pragma unroll
  for (int r = 0; r < 8; ++r) {
    int row = row0 + hf * 8 + r;
    if (row >= M) continue;
    float v = acc[r] + bc;
    if (do_relu && v < 0.f) v = 0.f;
    if (outb) outb[(size_t)row * out_ld + out_off + col] = (bf16)v;
    if (aout) {
      float mv = mult ? mult[row] : 1.f;
      if (mv != 0.f) atomicAdd(&aout[(size_t)sidx[row] * 128 + col], v * mv);
    }
  }
}

template <int NSEG>
__global__ __launch_bounds__(256) void gemm_bf16_wmma(
    const void* a0p, const int* a0i, int a0ld, int a0f,
    const void* a1p, const int* a1i, int a1ld, int a1f,
    const void* a2p, const int* a2i, int a2ld, int a2f,
    const bf16* __restrict__ Bt, const float* __restrict__ bias,
    int M, int do_relu,
    bf16* outb, int out_ld, int out_off,
    float* aout, const int* sidx, const float* mult) {
  constexpr int K = NSEG * 128;
  __shared__ bf16 ldsb[128 * kLdsLd];

  int tid   = threadIdx.x;
  int lane  = tid & 31;
  int wave  = tid >> 5;
  int rwave = wave & 3, cwave = wave >> 2;
  int rowb  = blockIdx.x * 128 + rwave * 32;
  int colb  = blockIdx.y * 128;
  int mrow  = lane & 15;
  int hf    = lane >> 4;

  int r0 = rowb + mrow;      if (r0 >= M) r0 = M - 1;
  int r1 = rowb + 16 + mrow; if (r1 >= M) r1 = M - 1;

  v8f zero = {0.f, 0.f, 0.f, 0.f, 0.f, 0.f, 0.f, 0.f};
  v8f acc[2][4];
#pragma unroll
  for (int rt = 0; rt < 2; ++rt)
#pragma unroll
    for (int ct = 0; ct < 4; ++ct) acc[rt][ct] = zero;

#pragma unroll
  for (int s = 0; s < NSEG; ++s) {
    // ---- stage B K-slab into LDS via async DMA (128 rows x 16 x 16B) ------
    __syncthreads();   // previous slab no longer in use by any wave
    {
      int c = tid;
#pragma unroll
      for (int it = 0; it < 8; ++it, c += 256) {
        int n = c >> 4, k = (c & 15) << 3;
        unsigned lofs = (unsigned)(size_t)&ldsb[n * kLdsLd + k];
        unsigned long long ga =
            (unsigned long long)&Bt[(size_t)(colb + n) * K + s * 128 + k];
        asm volatile("global_load_async_to_lds_b128 %0, %1, off"
                     :: "v"(lofs), "v"(ga) : "memory");
      }
    }
    asm volatile("s_wait_asynccnt 0x0" ::: "memory");
    __syncthreads();

    // -------- segment descriptor + row-gather resolution (hoisted) ---------
    const void* p; const int* ix; int ld, f32;
    if (s == 0)      { p = a0p; ix = a0i; ld = a0ld; f32 = a0f; }
    else if (s == 1) { p = a1p; ix = a1i; ld = a1ld; f32 = a1f; }
    else             { p = a2p; ix = a2i; ld = a2ld; f32 = a2f; }
    int sr0 = ix ? ix[r0] : r0;
    int sr1 = ix ? ix[r1] : r1;

#pragma unroll
    for (int kc = 0; kc < 128; kc += 32) {
      v16bf a0 = load_a_direct(p, ld, f32, sr0, kc, hf);
      v16bf a1 = load_a_direct(p, ld, f32, sr1, kc, hf);
      v16bf b0 = load_b_lds(ldsb, cwave * 64 +  0 + mrow, kc, hf);
      v16bf b1 = load_b_lds(ldsb, cwave * 64 + 16 + mrow, kc, hf);
      v16bf b2 = load_b_lds(ldsb, cwave * 64 + 32 + mrow, kc, hf);
      v16bf b3 = load_b_lds(ldsb, cwave * 64 + 48 + mrow, kc, hf);
      acc[0][0] = __builtin_amdgcn_wmma_f32_16x16x32_bf16(false, a0, false, b0, (short)0, acc[0][0], false, false);
      acc[0][1] = __builtin_amdgcn_wmma_f32_16x16x32_bf16(false, a0, false, b1, (short)0, acc[0][1], false, false);
      acc[0][2] = __builtin_amdgcn_wmma_f32_16x16x32_bf16(false, a0, false, b2, (short)0, acc[0][2], false, false);
      acc[0][3] = __builtin_amdgcn_wmma_f32_16x16x32_bf16(false, a0, false, b3, (short)0, acc[0][3], false, false);
      acc[1][0] = __builtin_amdgcn_wmma_f32_16x16x32_bf16(false, a1, false, b0, (short)0, acc[1][0], false, false);
      acc[1][1] = __builtin_amdgcn_wmma_f32_16x16x32_bf16(false, a1, false, b1, (short)0, acc[1][1], false, false);
      acc[1][2] = __builtin_amdgcn_wmma_f32_16x16x32_bf16(false, a1, false, b2, (short)0, acc[1][2], false, false);
      acc[1][3] = __builtin_amdgcn_wmma_f32_16x16x32_bf16(false, a1, false, b3, (short)0, acc[1][3], false, false);
    }
  }

#pragma unroll
  for (int rt = 0; rt < 2; ++rt)
#pragma unroll
    for (int ct = 0; ct < 4; ++ct)
      epilogue_tile(acc[rt][ct], rowb + rt * 16, hf,
                    colb + cwave * 64 + ct * 16 + mrow,
                    M, bias, do_relu, outb, out_ld, out_off, aout, sidx, mult);
}

// --------------------------- small support kernels -------------------------

__global__ void k_transpose_w(const float* __restrict__ W, bf16* __restrict__ Bt,
                              int K, int Nout) {
  int i = blockIdx.x * blockDim.x + threadIdx.x;
  if (i >= K * Nout) return;
  int n = i % Nout, k = i / Nout;
  Bt[(size_t)n * K + k] = (bf16)W[i];
}

__global__ void k_encode(const float* __restrict__ feat, const float* __restrict__ Wrow,
                         const float* __restrict__ b, bf16* __restrict__ out, int M) {
  int i = blockIdx.x * blockDim.x + threadIdx.x;
  if (i >= M * 128) return;
  int d = i & 127, m = i >> 7;
  out[i] = (bf16)(feat[m] * Wrow[d] + b[d]);
}

__global__ __launch_bounds__(64) void k_transform(const bf16* __restrict__ Sblk,
    const float* __restrict__ Wt1, const float* __restrict__ bt1,
    const float* __restrict__ Wt2, const float* __restrict__ bt2,
    float* __restrict__ mq, float* __restrict__ mc) {
  int tid = blockIdx.x * blockDim.x + threadIdx.x;
  if (tid >= kB * 40) return;
  int b = tid / 40, rem = tid % 40, side = rem / 20, slot = rem % 20;
  const bf16* p = Sblk + ((size_t)(b * 36 + side * 18 + slot)) * 768 + 640;
  float hid[16];
  for (int t = 0; t < 16; ++t) {
    float s = bt1[t];
    if (slot < 18)
      for (int d = 0; d < 128; ++d) s += (float)p[d] * Wt1[d * 16 + t];
    hid[t] = s > 0.f ? s : 0.f;
  }
  int lim = (side == 0) ? 15 : 18;   // qmask: q<15; cmask: c<18
  float* dst = (side == 0 ? mq : mc) + (size_t)b * 320 + slot * 16;
  for (int t2 = 0; t2 < 16; ++t2) {
    float s = bt2[t2];
    for (int t = 0; t < 16; ++t) s += hid[t] * Wt2[t * 16 + t2];
    dst[t2] = (slot < lim) ? s : 0.f;
  }
}

__global__ __launch_bounds__(32) void k_sinkhorn(const float* __restrict__ mq,
                                                 const float* __restrict__ mc,
                                                 float* __restrict__ plan) {
  int b = blockIdx.x, lane = threadIdx.x;
  __shared__ float la[20][21];
  if (lane < 20) {
    const float* qr = mq + (size_t)b * 320 + lane * 16;
    for (int c = 0; c < 20; ++c) {
      const float* cr = mc + (size_t)b * 320 + c * 16;
      float s = 0.f;
      for (int t = 0; t < 16; ++t) s += qr[t] * cr[t];
      la[lane][c] = s * 10.f;   // / SINKHORN_TEMP (0.1)
    }
  }
  __syncthreads();
  for (int it = 0; it < 10; ++it) {
    if (lane < 20) {
      float m = -1e30f;
      for (int c = 0; c < 20; ++c) m = fmaxf(m, la[lane][c]);
      float s = 0.f;
      for (int c = 0; c < 20; ++c) s += __expf(la[lane][c] - m);
      float l = m + __logf(s);
      for (int c = 0; c < 20; ++c) la[lane][c] -= l;
    }
    __syncthreads();
    if (lane < 20) {
      float m = -1e30f;
      for (int q = 0; q < 20; ++q) m = fmaxf(m, la[q][lane]);
      float s = 0.f;
      for (int q = 0; q < 20; ++q) s += __expf(la[q][lane] - m);
      float l = m + __logf(s);
      for (int q = 0; q < 20; ++q) la[q][lane] -= l;
    }
    __syncthreads();
  }
  if (lane < 20)
    for (int c = 0; c < 20; ++c)
      plan[(size_t)b * 400 + lane * 20 + c] = __expf(la[lane][c]);
}

__global__ __launch_bounds__(256) void k_plan_apply(const float* __restrict__ plan,
                                                    const bf16* __restrict__ Sblk,
                                                    bf16* __restrict__ Sout) {
  int n = blockIdx.x;
  int b = n / 36, s = n % 36;
  __shared__ float pc[18];
  int t = threadIdx.x;
  if (t < 18) {
    if (s < 18) pc[t] = plan[(size_t)b * 400 + s * 20 + t];
    else        pc[t] = plan[(size_t)b * 400 + t * 20 + (s - 18)];
  }
  __syncthreads();
  int srcbase = (s < 18) ? (b * 36 + 18) : (b * 36);
  for (int f = t; f < 768; f += 256) {
    float acc = 0.f;
    for (int c = 0; c < 18; ++c)
      acc += pc[c] * (float)Sblk[((size_t)(srcbase + c)) * 768 + f];
    Sout[(size_t)n * 768 + f] = (bf16)acc;
  }
}

__global__ __launch_bounds__(128) void k_score(const float* __restrict__ mq,
                                               const float* __restrict__ mc,
                                               const float* __restrict__ plan,
                                               float* __restrict__ out) {
  int b = blockIdx.x, t = threadIdx.x;
  __shared__ float red[128];
  float acc = 0.f;
  for (int i = t; i < 320; i += 128) {
    int q = i >> 4, tt = i & 15;
    float dot = 0.f;
    for (int c = 0; c < 20; ++c)
      dot += plan[(size_t)b * 400 + q * 20 + c] * mc[(size_t)b * 320 + c * 16 + tt];
    float v = mq[(size_t)b * 320 + q * 16 + tt] - dot;
    if (v > 0.f) acc += v;
  }
  red[t] = acc;
  __syncthreads();
  for (int off = 64; off > 0; off >>= 1) {
    if (t < off) red[t] += red[t + off];
    __syncthreads();
  }
  if (t == 0) out[b] = -red[0];
}

// ------------------------------- launcher ----------------------------------

extern "C" void kernel_launch(void* const* d_in, const int* in_sizes, int n_in,
                              void* d_out, int out_size, void* d_ws, size_t ws_size,
                              hipStream_t stream) {
  const float* node_features = (const float*)d_in[0];
  const float* edge_features = (const float*)d_in[1];
  const float* mult_factor   = (const float*)d_in[2];
  const float* W_ne = (const float*)d_in[3];  const float* b_ne = (const float*)d_in[4];
  const float* W_ee = (const float*)d_in[5];  const float* b_ee = (const float*)d_in[6];
  const float* W_m1 = (const float*)d_in[7];  const float* b_m1 = (const float*)d_in[8];
  const float* W_m2 = (const float*)d_in[9];  const float* b_m2 = (const float*)d_in[10];
  const float* W_u1 = (const float*)d_in[11]; const float* b_u1 = (const float*)d_in[12];
  const float* W_u2 = (const float*)d_in[13]; const float* b_u2 = (const float*)d_in[14];
  const float* W_c1 = (const float*)d_in[15]; const float* b_c1 = (const float*)d_in[16];
  const float* W_c2 = (const float*)d_in[17]; const float* b_c2 = (const float*)d_in[18];
  const float* W_t1 = (const float*)d_in[19]; const float* b_t1 = (const float*)d_in[20];
  const float* W_t2 = (const float*)d_in[21]; const float* b_t2 = (const float*)d_in[22];
  const int* from_idx = (const int*)d_in[23];
  const int* to_idx   = (const int*)d_in[24];
  float* outp = (float*)d_out;
  (void)in_sizes; (void)n_in; (void)out_size; (void)ws_size;

  char* wptr = (char*)d_ws;
  auto carve = [&](size_t bytes) -> char* {
    char* p = wptr;
    wptr += (bytes + 255) & ~size_t(255);
    return p;
  };
  bf16* Wc1t = (bf16*)carve((size_t)256 * 256 * 2);
  bf16* Wc2t = (bf16*)carve((size_t)128 * 256 * 2);
  bf16* Wm1t = (bf16*)carve((size_t)256 * 384 * 2);
  bf16* Wm2t = (bf16*)carve((size_t)128 * 256 * 2);
  bf16* Wu1t = (bf16*)carve((size_t)256 * 256 * 2);
  bf16* Wu2t = (bf16*)carve((size_t)128 * 256 * 2);
  bf16* nf_enc = (bf16*)carve((size_t)kN * 128 * 2);
  bf16* ef_enc = (bf16*)carve((size_t)kE * 128 * 2);
  bf16* Sin  = (bf16*)carve((size_t)kN * 768 * 2);
  bf16* Sblk = (bf16*)carve((size_t)kN * 768 * 2);
  bf16* hid  = (bf16*)carve((size_t)kE * 256 * 2);
  bf16* comb = (bf16*)carve((size_t)kN * 128 * 2);
  float* agg = (float*)carve((size_t)kN * 128 * 4);
  float* mq  = (float*)carve((size_t)kB * 320 * 4);
  float* mc  = (float*)carve((size_t)kB * 320 * 4);
  float* plan = (float*)carve((size_t)kB * 400 * 4);

  auto gemm = [&](int M, int K, int Nout,
                  const void* a0p, const int* a0i, int a0ld, int a0f,
                  const void* a1p, const int* a1i, int a1ld, int a1f,
                  const void* a2p, const int* a2i, int a2ld, int a2f,
                  const bf16* Bt, const float* bias, int do_relu,
                  bf16* outb, int out_ld, int out_off,
                  float* aout, const int* sidx, const float* mult) {
    dim3 grid((M + 127) / 128, Nout / 128);
    if (K == 256)
      gemm_bf16_wmma<2><<<grid, dim3(256), 0, stream>>>(
          a0p, a0i, a0ld, a0f, a1p, a1i, a1ld, a1f, a2p, a2i, a2ld, a2f,
          Bt, bias, M, do_relu, outb, out_ld, out_off, aout, sidx, mult);
    else
      gemm_bf16_wmma<3><<<grid, dim3(256), 0, stream>>>(
          a0p, a0i, a0ld, a0f, a1p, a1i, a1ld, a1f, a2p, a2i, a2ld, a2f,
          Bt, bias, M, do_relu, outb, out_ld, out_off, aout, sidx, mult);
  };
  auto tw = [&](const float* W, bf16* Bt, int K, int Nout) {
    int n = K * Nout;
    k_transpose_w<<<(n + 255) / 256, 256, 0, stream>>>(W, Bt, K, Nout);
  };

  // one-time prep
  tw(W_c1, Wc1t, 256, 256); tw(W_c2, Wc2t, 256, 128);
  tw(W_m1, Wm1t, 384, 256); tw(W_m2, Wm2t, 256, 128);
  tw(W_u1, Wu1t, 256, 256); tw(W_u2, Wu2t, 256, 128);
  k_encode<<<(kN * 128 + 255) / 256, 256, 0, stream>>>(node_features, W_ne, b_ne, nf_enc, kN);
  k_encode<<<(kE * 128 + 255) / 256, 256, 0, stream>>>(edge_features, W_ee, b_ee, ef_enc, kE);
  hipMemsetAsync(Sin, 0, (size_t)kN * 768 * 2, stream);

  for (int t = 0; t < 3; ++t) {
    hipMemsetAsync(Sblk, 0, (size_t)kN * 768 * 2, stream);  // block 0 stays zero
    for (int p = 1; p <= 5; ++p) {
      const bf16* hsrc = (p == 1) ? nf_enc : (Sblk + (size_t)(p - 1) * 128);
      int hld = (p == 1) ? 128 : 768;
      // comb = relu([h, inter] @ Wc1 + b) @ Wc2 + b
      gemm(kN, 256, 256, hsrc, nullptr, hld, 0,
           Sin + (size_t)(p - 1) * 128, nullptr, 768, 0,
           nullptr, nullptr, 0, 0,
           Wc1t, b_c1, 1, hid, 256, 0, nullptr, nullptr, nullptr);
      gemm(kN, 256, 128, hid, nullptr, 256, 0, hid + 128, nullptr, 256, 0,
           nullptr, nullptr, 0, 0,
           Wc2t, b_c2, 0, comb, 128, 0, nullptr, nullptr, nullptr);
      // messages: gather endpoints, 2-layer MLP, *mult, segment-sum via atomics
      hipMemsetAsync(agg, 0, (size_t)kN * 128 * 4, stream);
      gemm(kE, 384, 256, comb, from_idx, 128, 0, comb, to_idx, 128, 0,
           ef_enc, nullptr, 128, 0,
           Wm1t, b_m1, 1, hid, 256, 0, nullptr, nullptr, nullptr);
      gemm(kE, 256, 128, hid, nullptr, 256, 0, hid + 128, nullptr, 256, 0,
           nullptr, nullptr, 0, 0,
           Wm2t, b_m2, 0, nullptr, 0, 0, agg, to_idx, mult_factor);
      // update: h = relu([comb, agg] @ Wu1 + b) @ Wu2 + b -> Sblk slice p
      gemm(kN, 256, 256, comb, nullptr, 128, 0, agg, nullptr, 128, 1,
           nullptr, nullptr, 0, 0,
           Wu1t, b_u1, 1, hid, 256, 0, nullptr, nullptr, nullptr);
      gemm(kN, 256, 128, hid, nullptr, 256, 0, hid + 128, nullptr, 256, 0,
           nullptr, nullptr, 0, 0,
           Wu2t, b_u2, 0, Sblk, 768, p * 128, nullptr, nullptr, nullptr);
    }
    k_transform<<<(kB * 40 + 63) / 64, 64, 0, stream>>>(Sblk, W_t1, b_t1, W_t2, b_t2, mq, mc);
    k_sinkhorn<<<kB, 32, 0, stream>>>(mq, mc, plan);
    k_plan_apply<<<kN, 256, 0, stream>>>(plan, Sblk, Sin);
  }
  k_score<<<kB, 128, 0, stream>>>(mq, mc, plan, outp);
}
